// DeformableTransformer_72172630442096
// MI455X (gfx1250) — compile-verified, matrix operand order checked by reading, not run on hardware
//
#include <hip/hip_runtime.h>
#include <hip/hip_bf16.h>

// BEVFormer-style deformable transformer decoder for MI455X (gfx1250).
// All GEMMs in bf16 via v_wmma_f32_16x16x32_bf16 (f32 accum); flash-style
// self-attention (DH=32 == WMMA K) avoids the 134MB score tensor; the value
// projection stores a transposed bf16 V so the flash P@V operand is loaded
// with coalesced b128s; bf16 value cache for the deformable gather.
// GEMM keeps a single fragment set per iteration: at ~7 waves/SIMD the
// wave-level interleave hides load latency without spilling accumulators
// (manual double-buffering was tried and caused scratch spills per WMMA).

#define BS_  4
#define NQ_  1024
#define D_   256
#define NH_  8
#define NP_  4
#define L_   6
#define DFF_ 1024
#define HBEV 200
#define WBEV 150
#define NV_  (HBEV * WBEV)
#define DH_  32

typedef __attribute__((ext_vector_type(16))) __bf16 v16bf;
typedef __attribute__((ext_vector_type(8)))  float  v8f;

union FragBF { v16bf v; unsigned short h[16]; unsigned u[8]; };

__device__ __forceinline__ unsigned short f2bf(float f) {
  unsigned u = __float_as_uint(f);
  u += 0x7FFFu + ((u >> 16) & 1u);           // round-to-nearest-even
  return (unsigned short)(u >> 16);
}
__device__ __forceinline__ float bf2f(unsigned short h) {
  return __uint_as_float(((unsigned)h) << 16);
}

__device__ __forceinline__ v8f wmma_bf16(const FragBF& a, const FragBF& b, v8f c) {
  return __builtin_amdgcn_wmma_f32_16x16x32_bf16(false, a.v, false, b.v,
                                                 (short)0, c, false, false);
}

// Operand fragment where index n (A: row m / B^T: output col n) selects a
// row of a row-major [rows x ld] bf16 matrix with contiguous K.
// ISA 16-bit operand layout: lane = n%16, half = lane/16;
// VGPR j holds K = (j<4?0:16) + half*8 + (j&3)*2 .. +1.
__device__ __forceinline__ FragBF load_frag_nmajor(const unsigned short* __restrict__ P,
                                                   long ld, long row0, long col0,
                                                   long maxRow) {
  int lane = (int)(threadIdx.x & 31u);
  long r = row0 + (lane & 15);
  if (r >= maxRow) r = maxRow - 1;            // clamp (stores are guarded)
  int hh = lane >> 4;
  const unsigned short* p = P + r * ld + col0;
  FragBF f;
#pragma unroll
  for (int j = 0; j < 8; ++j) {
    int kb = ((j & 4) ? 16 : 0) + hh * 8 + (j & 3) * 2;
    f.u[j] = *(const unsigned*)(p + kb);      // 2 bf16 per dword -> b128 pairs
  }
  return f;
}

// ---------------------------------------------------------------------------
// Generic GEMM: C[M x N] = A[M x K](bf16) @ B[N x K]^T(bf16) + bias
// flags: bit0 = relu, bit1 = bf16 output, bit2 = bf16 output transposed into
//        attention V^T layout [b, h, dh, nq] (requires N==D_, M==BS_*NQ_).
// Block = 256 threads (8 waves, 4x2), tile 128(M) x 128(N), wave tile 32x64:
// 8 WMMAs per 6 fragment loads. Single fragment set (no spills); latency is
// hidden by multi-wave occupancy and cache prefetch two K-tiles ahead.
// ---------------------------------------------------------------------------
__global__ void __launch_bounds__(256, 2)
k_gemm_wmma(const unsigned short* __restrict__ A, long lda,
            const unsigned short* __restrict__ B, long ldb,
            const float* __restrict__ bias,
            void* __restrict__ Cout, long ldc,
            long M, long N, long K, int flags) {
  int wave = (int)(threadIdx.x >> 5);
  int wm = wave & 3, wn = wave >> 2;
  long row0 = (long)blockIdx.x * 128 + (long)wm * 32;
  long col0 = (long)blockIdx.y * 128 + (long)wn * 64;
  long pa = (row0 < M) ? row0 : (M - 1);
  long pb = (col0 < N) ? col0 : (N - 1);

  v8f acc[2][4] = {};
  for (long k = 0; k < K; k += 32) {
    FragBF a0 = load_frag_nmajor(A, lda, row0,      k, M);
    FragBF a1 = load_frag_nmajor(A, lda, row0 + 16, k, M);
    FragBF b0 = load_frag_nmajor(B, ldb, col0,      k, N);
    FragBF b1 = load_frag_nmajor(B, ldb, col0 + 16, k, N);
    FragBF b2 = load_frag_nmajor(B, ldb, col0 + 32, k, N);
    FragBF b3 = load_frag_nmajor(B, ldb, col0 + 48, k, N);
    if (k + 64 < K) {                          // warm the tile two steps ahead
      __builtin_prefetch((const void*)(A + pa * lda + k + 64), 0, 3);
      __builtin_prefetch((const void*)(B + pb * ldb + k + 64), 0, 3);
    }
    acc[0][0] = wmma_bf16(a0, b0, acc[0][0]);
    acc[0][1] = wmma_bf16(a0, b1, acc[0][1]);
    acc[0][2] = wmma_bf16(a0, b2, acc[0][2]);
    acc[0][3] = wmma_bf16(a0, b3, acc[0][3]);
    acc[1][0] = wmma_bf16(a1, b0, acc[1][0]);
    acc[1][1] = wmma_bf16(a1, b1, acc[1][1]);
    acc[1][2] = wmma_bf16(a1, b2, acc[1][2]);
    acc[1][3] = wmma_bf16(a1, b3, acc[1][3]);
  }

  int lane = (int)(threadIdx.x & 31u);
  int hh = lane >> 4, cn = lane & 15;
#pragma unroll
  for (int mi = 0; mi < 2; ++mi)
#pragma unroll
    for (int ni = 0; ni < 4; ++ni)
#pragma unroll
      for (int r = 0; r < 8; ++r) {
        long rr = row0 + mi * 16 + r + hh * 8;   // C layout: VGPR r, half hh
        long cc = col0 + ni * 16 + cn;
        if (rr < M && cc < N) {
          float v = acc[mi][ni][r];
          if (bias) v += bias[cc];
          if (flags & 1) v = fmaxf(v, 0.0f);
          if (flags & 4) {
            // V^T layout: [b, h, dh, nq]
            long bb = rr / NQ_, qq = rr % NQ_;
            long hd = cc / DH_, dd = cc % DH_;
            ((unsigned short*)Cout)[((bb * NH_ + hd) * DH_ + dd) * NQ_ + qq] = f2bf(v);
          } else if (flags & 2) {
            ((unsigned short*)Cout)[rr * ldc + cc] = f2bf(v);
          } else {
            ((float*)Cout)[rr * ldc + cc] = v;
          }
        }
      }
}

// ---------------------------------------------------------------------------
// Flash self-attention: one wave owns a 16-query tile of one (batch, head).
// DH=32 == WMMA K: QK^T is a single WMMA per 16-key tile; streams 32 keys
// per iteration with online softmax; P tile goes through 1KB LDS to convert
// C-layout -> A-operand layout. V is pre-transposed ([b,h,dh,nq]) so both
// operands of P@V load with coalesced contiguous-K fragments. V fragments
// and the next iteration's K fragments are loaded before the softmax VALU
// block so global latency hides behind ~100 VALU ops.
// ---------------------------------------------------------------------------
__global__ void __launch_bounds__(32)
k_flash_sa(const unsigned short* __restrict__ Q, long ldq,
           const unsigned short* __restrict__ Kv, long ldk,
           const unsigned short* __restrict__ VT,
           unsigned short* __restrict__ O, long ldo,
           int nq, int nk, float scale) {
  __shared__ unsigned short pshare[16 * 32];
  int b = blockIdx.z, h = blockIdx.y;
  const unsigned short* q   = Q  + (long)b * nq * ldq + h * DH_;
  const unsigned short* kp  = Kv + (long)b * nk * ldk + h * DH_;
  const unsigned short* vtp = VT + ((long)b * NH_ + h) * DH_ * (long)nk;
  unsigned short*       op  = O  + (long)b * nq * ldo + h * DH_;
  long row0 = (long)blockIdx.x * 16;

  int lane = (int)(threadIdx.x & 31u);
  int hh = lane >> 4, cn = lane & 15;

  FragBF qf = load_frag_nmajor(q, ldq, row0, 0, nq);
  v8f o0 = {}, o1 = {};
  float mrow[8], lrow[8];
#pragma unroll
  for (int r = 0; r < 8; ++r) { mrow[r] = -3.0e38f; lrow[r] = 0.0f; }

  FragBF kf0 = load_frag_nmajor(kp, ldk, 0,  0, nk);
  FragBF kf1 = load_frag_nmajor(kp, ldk, 16, 0, nk);

  for (int k0 = 0; k0 < nk; k0 += 32) {
    // independent loads first: V for this tile
    FragBF vf0 = load_frag_nmajor(vtp, nk, 0,  k0, DH_);      // V^T: dh 0..15
    FragBF vf1 = load_frag_nmajor(vtp, nk, 16, k0, DH_);      // V^T: dh 16..31

    v8f z = {};
    v8f s0 = wmma_bf16(qf, kf0, z);     // scores, keys k0..k0+15
    v8f s1 = wmma_bf16(qf, kf1, z);     // scores, keys k0+16..k0+31

    // preload next K tile; latency hides behind the softmax below
    if (k0 + 32 < nk) {
      kf0 = load_frag_nmajor(kp, ldk, k0 + 32, 0, nk);
      kf1 = load_frag_nmajor(kp, ldk, k0 + 48, 0, nk);
    }

#pragma unroll
    for (int r = 0; r < 8; ++r) {
      float a = s0[r] * scale;
      float c = s1[r] * scale;
      float mx = fmaxf(a, c);
#pragma unroll
      for (int o = 1; o < 16; o <<= 1)  // row-max across the 16 N lanes
        mx = fmaxf(mx, __shfl_xor(mx, o, 32));
      float nm   = fmaxf(mrow[r], mx);
      float corr = __expf(mrow[r] - nm);
      float p0 = __expf(a - nm);
      float p1 = __expf(c - nm);
      float ps = p0 + p1;
#pragma unroll
      for (int o = 1; o < 16; o <<= 1)  // row-sum
        ps += __shfl_xor(ps, o, 32);
      lrow[r] = lrow[r] * corr + ps;
      mrow[r] = nm;
      o0[r] *= corr;
      o1[r] *= corr;
      int prow = r + hh * 8;            // C layout -> row-major LDS tile
      pshare[prow * 32 + cn]      = f2bf(p0);
      pshare[prow * 32 + cn + 16] = f2bf(p1);
    }
    __syncthreads();
    FragBF pf = load_frag_nmajor(pshare, 32, 0, 0, 16);       // A: 16 q x 32 k
    o0 = wmma_bf16(pf, vf0, o0);
    o1 = wmma_bf16(pf, vf1, o1);
    __syncthreads();
  }

#pragma unroll
  for (int r = 0; r < 8; ++r) {
    long rq = row0 + r + hh * 8;
    float inv = 1.0f / lrow[r];
    op[rq * ldo + cn]      = f2bf(o0[r] * inv);
    op[rq * ldo + cn + 16] = f2bf(o1[r] * inv);
  }
}

// ---------------------------------------------------------------------------
// Fused residual + LayerNorm; optionally emits bf16(x), bf16(x+pos), hs copy.
// One block (256 threads) per row, D=256 -> one channel per thread.
// ---------------------------------------------------------------------------
__global__ void k_ln(const float* __restrict__ x, const float* __restrict__ t,
                     const float* __restrict__ w, const float* __restrict__ b,
                     const float* __restrict__ pos,
                     float* __restrict__ xo, unsigned short* __restrict__ xbf,
                     unsigned short* __restrict__ xposbf, float* __restrict__ hs) {
  __shared__ float r1[8], r2[8];
  long row = blockIdx.x;
  int c = (int)threadIdx.x;
  long idx = row * D_ + c;
  float v = x[idx] + t[idx];
  float s1 = v, s2 = v * v;
#pragma unroll
  for (int o = 1; o < 32; o <<= 1) {
    s1 += __shfl_xor(s1, o, 32);
    s2 += __shfl_xor(s2, o, 32);
  }
  int wv = (int)(threadIdx.x >> 5), ln = (int)(threadIdx.x & 31u);
  if (ln == 0) { r1[wv] = s1; r2[wv] = s2; }
  __syncthreads();
  if (wv == 0) {
    float a = (ln < 8) ? r1[ln] : 0.0f;
    float q = (ln < 8) ? r2[ln] : 0.0f;
#pragma unroll
    for (int o = 1; o < 8; o <<= 1) { a += __shfl_xor(a, o, 32); q += __shfl_xor(q, o, 32); }
    if (ln == 0) { r1[0] = a; r2[0] = q; }
  }
  __syncthreads();
  float mean = r1[0] * (1.0f / D_);
  float var  = r2[0] * (1.0f / D_) - mean * mean;
  float y = (v - mean) * rsqrtf(var + 1e-5f) * w[c] + b[c];
  xo[idx] = y;
  if (xbf)    xbf[idx]    = f2bf(y);
  if (xposbf) xposbf[idx] = f2bf(y + pos[idx]);
  if (hs)     hs[idx]     = y;
}

// ---------------------------------------------------------------------------
// Deformable sampling: 1 wave per (b, q, h); lane = channel (DH=32).
// Softmax over NP=4 points, bilinear gather with zero padding from bf16 V.
// ---------------------------------------------------------------------------
__global__ void k_msda(const float* __restrict__ ref, const float* __restrict__ off,
                       const float* __restrict__ aw,
                       const unsigned short* __restrict__ vsrc,
                       unsigned short* __restrict__ outb) {
  int b = blockIdx.z, h = blockIdx.y, qi = blockIdx.x;
  int lane = (int)threadIdx.x;
  long row = (long)b * NQ_ + qi;
  float rx = ref[row * 2 + 0];
  float ry = ref[row * 2 + 1];
  const float* offp = off + row * (NH_ * NP_ * 2) + h * (NP_ * 2);
  const float* awp  = aw  + row * (NH_ * NP_)     + h * NP_;

  float a0 = awp[0], a1 = awp[1], a2 = awp[2], a3 = awp[3];
  float am = fmaxf(fmaxf(a0, a1), fmaxf(a2, a3));
  float e0 = __expf(a0 - am), e1 = __expf(a1 - am);
  float e2 = __expf(a2 - am), e3 = __expf(a3 - am);
  float inv = 1.0f / (e0 + e1 + e2 + e3);
  float wgt[4] = { e0 * inv, e1 * inv, e2 * inv, e3 * inv };

  float acc = 0.0f;
  const unsigned short* vb = vsrc + (long)b * NV_ * D_ + h * DH_ + lane;
#pragma unroll
  for (int p = 0; p < NP_; ++p) {
    // (ref + off/norm)*size - 0.5  ==  ref*size + off - 0.5
    float xx = rx * (float)WBEV + offp[p * 2 + 0] - 0.5f;
    float yy = ry * (float)HBEV + offp[p * 2 + 1] - 0.5f;
    float x0 = floorf(xx), y0 = floorf(yy);
#pragma unroll
    for (int c = 0; c < 4; ++c) {
      float xi = x0 + (float)(c & 1);
      float yi = y0 + (float)(c >> 1);
      float w = (1.0f - fabsf(xx - xi)) * (1.0f - fabsf(yy - yi));
      bool valid = (xi >= 0.0f) && (xi < (float)WBEV) &&
                   (yi >= 0.0f) && (yi < (float)HBEV);
      if (valid) {
        long idx = (long)yi * WBEV + (long)xi;
        acc += bf2f(vb[idx * D_]) * (w * wgt[p]);
      }
    }
  }
  outb[row * D_ + h * DH_ + lane] = f2bf(acc);
}

// --------------------------- small utility kernels -------------------------
__global__ void k_f32_to_bf16(const float* __restrict__ in,
                              unsigned short* __restrict__ out, long n) {
  long i = (long)blockIdx.x * blockDim.x + threadIdx.x;
  if (i < n) out[i] = f2bf(in[i]);
}

__global__ void k_prep(const float* __restrict__ q, const float* __restrict__ pos,
                       float* __restrict__ xo, unsigned short* __restrict__ xbf,
                       unsigned short* __restrict__ xposbf, long n) {
  long i = (long)blockIdx.x * blockDim.x + threadIdx.x;
  if (i < n) {
    float v = q[i];
    xo[i] = v;
    xbf[i] = f2bf(v);
    xposbf[i] = f2bf(v + pos[i]);
  }
}

__global__ void k_refs(const float* __restrict__ ref, float* __restrict__ out, long n) {
  long i = (long)blockIdx.x * blockDim.x + threadIdx.x;
  if (i < n) {
    float v = ref[i];
#pragma unroll
    for (int l = 0; l < L_; ++l) out[(long)l * n + i] = v;
  }
}

// ---------------------------------------------------------------------------
extern "C" void kernel_launch(void* const* d_in, const int* in_sizes, int n_in,
                              void* d_out, int out_size, void* d_ws, size_t ws_size,
                              hipStream_t stream) {
  (void)in_sizes; (void)n_in; (void)out_size; (void)ws_size;
  const float* query     = (const float*)d_in[0];
  const float* query_pos = (const float*)d_in[1];
  const float* refpts    = (const float*)d_in[2];
  const float* src       = (const float*)d_in[3];
  const float* sa_in_w   = (const float*)d_in[6];
  const float* sa_in_b   = (const float*)d_in[7];
  const float* sa_out_w  = (const float*)d_in[8];
  const float* sa_out_b  = (const float*)d_in[9];
  const float* ca_off_w  = (const float*)d_in[10];
  const float* ca_off_b  = (const float*)d_in[11];
  const float* ca_aw_w   = (const float*)d_in[12];
  const float* ca_aw_b   = (const float*)d_in[13];
  const float* ca_val_w  = (const float*)d_in[14];
  const float* ca_val_b  = (const float*)d_in[15];
  const float* ca_out_w  = (const float*)d_in[16];
  const float* ca_out_b  = (const float*)d_in[17];
  const float* n1_w = (const float*)d_in[18];
  const float* n1_b = (const float*)d_in[19];
  const float* n2_w = (const float*)d_in[20];
  const float* n2_b = (const float*)d_in[21];
  const float* n3_w = (const float*)d_in[22];
  const float* n3_b = (const float*)d_in[23];
  const float* ff1_w = (const float*)d_in[24];
  const float* ff1_b = (const float*)d_in[25];
  const float* ff2_w = (const float*)d_in[26];
  const float* ff2_b = (const float*)d_in[27];

  const long M  = (long)BS_ * NQ_;   // 4096 query rows
  const long MV = (long)BS_ * NV_;   // 120000 value rows

  // bump allocator over d_ws (~165 MB total)
  char* wsb = (char*)d_ws;
  size_t woff = 0;
  auto alloc = [&](size_t bytes) -> void* {
    void* p = wsb + woff;
    woff += (bytes + 255) & ~(size_t)255;
    return p;
  };

  unsigned short* w_sa_in  = (unsigned short*)alloc((size_t)L_ * 3 * D_ * D_ * 2);
  unsigned short* w_sa_out = (unsigned short*)alloc((size_t)L_ * D_ * D_ * 2);
  unsigned short* w_ca_off = (unsigned short*)alloc((size_t)L_ * 64 * D_ * 2);
  unsigned short* w_ca_aw  = (unsigned short*)alloc((size_t)L_ * 32 * D_ * 2);
  unsigned short* w_ca_val = (unsigned short*)alloc((size_t)L_ * D_ * D_ * 2);
  unsigned short* w_ca_out = (unsigned short*)alloc((size_t)L_ * D_ * D_ * 2);
  unsigned short* w_ff1    = (unsigned short*)alloc((size_t)L_ * DFF_ * D_ * 2);
  unsigned short* w_ff2    = (unsigned short*)alloc((size_t)L_ * D_ * DFF_ * 2);
  unsigned short* src_bf   = (unsigned short*)alloc((size_t)MV * D_ * 2);
  float*          x        = (float*)alloc((size_t)M * D_ * 4);
  unsigned short* x_bf     = (unsigned short*)alloc((size_t)M * D_ * 2);
  unsigned short* xpos_bf  = (unsigned short*)alloc((size_t)M * D_ * 2);
  unsigned short* skbuf    = (unsigned short*)alloc((size_t)M * 2 * D_ * 2); // q|k
  unsigned short* vtbuf    = (unsigned short*)alloc((size_t)M * D_ * 2);     // V^T
  unsigned short* abuf     = (unsigned short*)alloc((size_t)M * D_ * 2);
  float*          tmp      = (float*)alloc((size_t)M * D_ * 4);
  unsigned short* vsrc_bf  = (unsigned short*)alloc((size_t)MV * D_ * 2);
  float*          off_buf  = (float*)alloc((size_t)M * 64 * 4);
  float*          aw_buf   = (float*)alloc((size_t)M * 32 * 4);
  unsigned short* ff_bf    = (unsigned short*)alloc((size_t)M * DFF_ * 2);

  auto conv = [&](const float* s, unsigned short* dd, long n) {
    k_f32_to_bf16<<<dim3((unsigned)((n + 255) / 256)), dim3(256), 0, stream>>>(s, dd, n);
  };
  conv(sa_in_w,  w_sa_in,  (long)L_ * 3 * D_ * D_);
  conv(sa_out_w, w_sa_out, (long)L_ * D_ * D_);
  conv(ca_off_w, w_ca_off, (long)L_ * 64 * D_);
  conv(ca_aw_w,  w_ca_aw,  (long)L_ * 32 * D_);
  conv(ca_val_w, w_ca_val, (long)L_ * D_ * D_);
  conv(ca_out_w, w_ca_out, (long)L_ * D_ * D_);
  conv(ff1_w,    w_ff1,    (long)L_ * DFF_ * D_);
  conv(ff2_w,    w_ff2,    (long)L_ * D_ * DFF_);
  conv(src,      src_bf,   MV * D_);

  k_prep<<<dim3((unsigned)((M * D_ + 255) / 256)), dim3(256), 0, stream>>>(
      query, query_pos, x, x_bf, xpos_bf, M * D_);

  auto gemm = [&](const unsigned short* A, long lda, const unsigned short* B, long ldb,
                  const float* bias, void* C, long ldc, long Mm, long Nn, long Kk,
                  int flags) {
    dim3 g((unsigned)((Mm + 127) / 128), (unsigned)((Nn + 127) / 128), 1);
    k_gemm_wmma<<<g, dim3(256), 0, stream>>>(A, lda, B, ldb, bias, C, ldc, Mm, Nn, Kk, flags);
  };

  const float scale = 0.17677669529663687f;  // 1/sqrt(DH)

  for (int l = 0; l < L_; ++l) {
    const unsigned short* wsa = w_sa_in + (size_t)l * 3 * D_ * D_;
    // q,k from (out + pos); v from out (stored transposed for flash)
    gemm(xpos_bf, D_, wsa, D_, sa_in_b + (size_t)l * 3 * D_,
         skbuf, 2 * D_, M, 2 * D_, D_, 2);
    gemm(x_bf, D_, wsa + (size_t)2 * D_ * D_, D_, sa_in_b + (size_t)l * 3 * D_ + 2 * D_,
         vtbuf, D_, M, D_, D_, 4);
    // flash self-attention
    k_flash_sa<<<dim3(NQ_ / 16, NH_, BS_), dim3(32), 0, stream>>>(
        skbuf, 2 * D_, skbuf + D_, 2 * D_, vtbuf, abuf, D_, NQ_, NQ_, scale);
    gemm(abuf, D_, w_sa_out + (size_t)l * D_ * D_, D_, sa_out_b + (size_t)l * D_,
         tmp, D_, M, D_, D_, 0);
    // out = LN(out + sa; n2); emit bf16(out+pos) for CA projections
    k_ln<<<dim3((unsigned)M), dim3(256), 0, stream>>>(
        x, tmp, n2_w + (size_t)l * D_, n2_b + (size_t)l * D_, query_pos,
        x, (unsigned short*)nullptr, xpos_bf, (float*)nullptr);

    // CA: value projection over the whole BEV grid (dominant GEMM)
    gemm(src_bf, D_, w_ca_val + (size_t)l * D_ * D_, D_, ca_val_b + (size_t)l * D_,
         vsrc_bf, D_, MV, D_, D_, 2);
    gemm(xpos_bf, D_, w_ca_off + (size_t)l * 64 * D_, D_, ca_off_b + (size_t)l * 64,
         off_buf, 64, M, 64, D_, 0);
    gemm(xpos_bf, D_, w_ca_aw + (size_t)l * 32 * D_, D_, ca_aw_b + (size_t)l * 32,
         aw_buf, 32, M, 32, D_, 0);
    k_msda<<<dim3(NQ_, NH_, BS_), dim3(32), 0, stream>>>(
        refpts, off_buf, aw_buf, vsrc_bf, abuf);
    gemm(abuf, D_, w_ca_out + (size_t)l * D_ * D_, D_, ca_out_b + (size_t)l * D_,
         tmp, D_, M, D_, D_, 0);
    // out = LN(out + ca; n1); emit bf16(out) for FFN
    k_ln<<<dim3((unsigned)M), dim3(256), 0, stream>>>(
        x, tmp, n1_w + (size_t)l * D_, n1_b + (size_t)l * D_, query_pos,
        x, x_bf, (unsigned short*)nullptr, (float*)nullptr);

    // FFN with fused ReLU + bf16 intermediate
    gemm(x_bf, D_, w_ff1 + (size_t)l * DFF_ * D_, D_, ff1_b + (size_t)l * DFF_,
         ff_bf, DFF_, M, DFF_, D_, 1 | 2);
    gemm(ff_bf, DFF_, w_ff2 + (size_t)l * D_ * DFF_, DFF_, ff2_b + (size_t)l * D_,
         tmp, D_, M, D_, DFF_, 0);
    // out = LN(out + ff; n3); emit hs[l] and the next layer's bf16 inputs
    k_ln<<<dim3((unsigned)M), dim3(256), 0, stream>>>(
        x, tmp, n3_w + (size_t)l * D_, n3_b + (size_t)l * D_, query_pos,
        x, x_bf, xpos_bf, (float*)d_out + (size_t)l * M * D_);
  }

  // refs: L copies of reference_points after the hs block
  k_refs<<<dim3((unsigned)((M * 2 + 255) / 256)), dim3(256), 0, stream>>>(
      refpts, (float*)d_out + (size_t)L_ * M * D_, M * 2);
}
